// GraphMap_41283225649618
// MI455X (gfx1250) — matrix operand order repaired and checked by reading, not compiled, not run on hardware
//
#include <hip/hip_runtime.h>
#include <hip/hip_bf16.h>
#include <cstdint>
#include <cstddef>

typedef __attribute__((ext_vector_type(16))) _Float16 v16h;
typedef __attribute__((ext_vector_type(8)))  _Float16 v8h;
typedef __attribute__((ext_vector_type(8)))  float    v8f;
typedef _Float16 half_t;

#define N_NODES 10000
#define D_IN    784
#define D_PAD   800              /* 784 padded to multiple of 32 */
#define KM1     15
#define E_EDGES (N_NODES * KM1)
#define HID     128
#define M_TILES (N_NODES / 16)   /* 625, exact */

static inline int cdiv(int a, int b) { return (a + b - 1) / b; }

// ---------------------------------------------------------------------------
// Pack a row-major f32 weight matrix W[K, N] into the CDNA5 WMMA B-operand
// f16 register image: per tile nt / 32-deep k-block kb, (lane, half) holds
//   lanes 0-15 : K = 0..7, 16..23   (2 per VGPR), n = lane
//   lanes 16-31: K = 8..15, 24..31,               n = lane - 16
// kbOff/KBt interleave two K-segments per tile (implicit concat inputs).
// ---------------------------------------------------------------------------
__global__ void pack_b_kernel(const float* __restrict__ W, int Kreal, int Nreal,
                              int KB, int KBt, int kbOff, int NT,
                              half_t* __restrict__ dst)
{
  int idx = blockIdx.x * blockDim.x + threadIdx.x;
  int total = NT * KB * 512;
  if (idx >= total) return;
  int h    = idx & 15;
  int lane = (idx >> 4) & 31;
  int kb   = (idx >> 9) % KB;
  int nt   = idx / (KB * 512);
  int k = kb * 32 + (h & 7) + ((h >> 3) << 4) + ((lane >> 4) << 3);
  int n = nt * 16 + (lane & 15);
  float v = (k < Kreal && n < Nreal) ? W[(size_t)k * Nreal + n] : 0.0f;
  dst[(((size_t)(nt * KBt + kbOff + kb)) * 32 + lane) * 16 + h] = (half_t)v;
}

// x [N,784] f32 -> x_h [N,800] f16 with zero pad
__global__ void convert_pad_kernel(const float* __restrict__ src, half_t* __restrict__ dst,
                                   int nrows)
{
  int t = blockIdx.x * blockDim.x + threadIdx.x;
  if (t >= nrows * D_PAD) return;
  int n = t / D_PAD;
  int j = t - n * D_PAD;
  dst[t] = (j < D_IN) ? (half_t)src[(size_t)n * D_IN + j] : (half_t)0.0f;
}

// zero columns 784..799 of a [N,800] f16 buffer
__global__ void pad_cols_kernel(half_t* __restrict__ buf)
{
  int t = blockIdx.x * blockDim.x + threadIdx.x;
  if (t >= N_NODES * 16) return;
  int n = t >> 4;
  buf[(size_t)n * D_PAD + D_IN + (t & 15)] = (half_t)0.0f;
}

// ---------------------------------------------------------------------------
// One K-segment: A is f16 row-major (lda mult of 8, fully zero-padded), loaded
// straight into the 16-bit A-operand layout:
//   lane l (m=l&15), halves 0..7  <- A[m][k0 + base0 .. +7]
//                    halves 8..15 <- A[m][k0 + 16 + base0 .. +7], base0=8*(l>=16)
// Exactly TW output tiles share each A fragment -- no guards, no divergence.
// ---------------------------------------------------------------------------
template<int TW>
__device__ __forceinline__ void gemm_seg(const half_t* __restrict__ A, int lda, int KB,
                                         const half_t* __restrict__ bb, int& kbg, int KBt,
                                         int m, int base0, v8f acc[TW])
{
  const half_t* arow = A + (size_t)m * lda + base0;
  for (int kb = 0; kb < KB; ++kb, ++kbg) {
    const v8h lo = *(const v8h*)(arow + kb * 32);
    const v8h hi = *(const v8h*)(arow + kb * 32 + 16);
    const v16h a = __builtin_shufflevector(lo, hi, 0,1,2,3,4,5,6,7,8,9,10,11,12,13,14,15);
    #pragma unroll
    for (int t = 0; t < TW; ++t) {
      const v16h b = *(const v16h*)(bb + ((size_t)t * KBt + kbg) * 512);
      acc[t] = __builtin_amdgcn_wmma_f32_16x16x32_f16(
          false, a, false, b, (short)0, acc[t], false, false);
    }
  }
}

// C[M,Nout] = act( concatK(A0, A1) @ Bpk + bias )
// One wave32 computes 16 rows x TW 16-col tiles starting at tile ntOff + g*TW.
template<int TW>
__global__ void gemm_wmma_kernel(const half_t* __restrict__ A0, int lda0, int KB0,
                                 const half_t* __restrict__ A1, int lda1, int KB1,
                                 const half_t* __restrict__ Bpk,
                                 const float* __restrict__ bias,
                                 int Nout, int ntOff, int NTG,
                                 void* __restrict__ Cp, int ldc, int Mtiles,
                                 int relu, int out_f16)
{
  const int lane = threadIdx.x & 31;
  // wave-uniform by construction; readfirstlane moves it (and all derived
  // addressing) into SGPRs so every branch before the epilogue is scalar.
  const int wid = __builtin_amdgcn_readfirstlane(
      blockIdx.x * (blockDim.x >> 5) + (threadIdx.x >> 5));
  if (wid >= Mtiles * NTG) return;
  const int mt    = wid / NTG;
  const int nt0   = ntOff + (wid - mt * NTG) * TW;
  const int m     = mt * 16 + (lane & 15);
  const int base0 = (lane >> 4) << 3;
  const int KBt   = KB0 + KB1;
  const half_t* bb = Bpk + ((size_t)nt0 * KBt) * 512 + lane * 16;

  v8f acc[TW];
  const v8f zero = {0.f,0.f,0.f,0.f,0.f,0.f,0.f,0.f};
  #pragma unroll
  for (int t = 0; t < TW; ++t) acc[t] = zero;

  int kbg = 0;
  gemm_seg<TW>(A0, lda0, KB0, bb, kbg, KBt, m, base0, acc);
  if (A1) gemm_seg<TW>(A1, lda1, KB1, bb, kbg, KBt, m, base0, acc);

  const int rbase = mt * 16 + ((lane >> 4) << 3);  // C layout: vgpr r -> row r + 8*(lane>=16)
  #pragma unroll
  for (int t = 0; t < TW; ++t) {
    const int n = (nt0 + t) * 16 + (lane & 15);
    if (n < Nout) {                                // store-only divergence
      const float bv = bias ? bias[n] : 0.0f;
      #pragma unroll
      for (int r = 0; r < 8; ++r) {
        float v = acc[t][r] + bv;
        if (relu) v = fmaxf(v, 0.0f);
        const size_t off = (size_t)(rbase + r) * ldc + n;
        if (out_f16) ((half_t*)Cp)[off] = (half_t)v;
        else         ((float*)Cp)[off]  = v;
      }
    }
  }
}

// ------------------------------ edge / scatter ------------------------------
__global__ void count_src_kernel(const int* __restrict__ src, int* __restrict__ cnt, int ne)
{
  int t = blockIdx.x * blockDim.x + threadIdx.x;
  if (t < ne) atomicAdd(&cnt[src[t]], 1);
}

// hidden(e) = ReLU(base[src_e] + ea_e*w_ea + b); scatter-add into hsum[src_e]
// one thread = 4 channels (float4), E*32 threads; all buffers L2-resident
__global__ void edge_hidden_kernel(const float* __restrict__ base,
                                   const int* __restrict__ src,
                                   const float* __restrict__ ea,
                                   const float* __restrict__ w_ea,
                                   const float* __restrict__ b,
                                   float* __restrict__ hsum, int nedges)
{
  int t = blockIdx.x * blockDim.x + threadIdx.x;
  if (t >= nedges * 32) return;
  const int e = t >> 5;
  const int j = (t & 31) * 4;
  const int s = src[e];
  const float eav = ea[e];
  const float4 bs = *(const float4*)(base + (size_t)s * HID + j);
  const float4 wv = *(const float4*)(w_ea + j);
  const float4 bb = *(const float4*)(b + j);
  float* dst = hsum + (size_t)s * HID + j;
  atomicAdd(dst + 0, fmaxf(fmaf(eav, wv.x, bs.x + bb.x), 0.0f));
  atomicAdd(dst + 1, fmaxf(fmaf(eav, wv.y, bs.y + bb.y), 0.0f));
  atomicAdd(dst + 2, fmaxf(fmaf(eav, wv.z, bs.z + bb.z), 0.0f));
  atomicAdd(dst + 3, fmaxf(fmaf(eav, wv.w, bs.w + bb.w), 0.0f));
}

// hmean (f16) = cnt>0 ? hsum/cnt : 0
__global__ void mean_to_h_kernel(const float* __restrict__ hsum, const int* __restrict__ cnt,
                                 half_t* __restrict__ dst, int total)
{
  int t = blockIdx.x * blockDim.x + threadIdx.x;
  if (t >= total) return;
  int c = cnt[t >> 7];
  dst[t] = (c > 0) ? (half_t)(hsum[t] / (float)c) : (half_t)0.0f;
}

// scatter_mean semantics: rows with cnt==0 -> agg row = 0 (not bias)
__global__ void zero_rows_h_kernel(half_t* __restrict__ buf, int ld, int ncols,
                                   const int* __restrict__ cnt, int nrows)
{
  long long t = (long long)blockIdx.x * blockDim.x + threadIdx.x;
  if (t >= (long long)nrows * ncols) return;
  int n = (int)(t / ncols);
  int j = (int)(t - (long long)n * ncols);
  if (cnt[n] == 0) buf[(size_t)n * ld + j] = (half_t)0.0f;
}

__global__ void dist_kernel(const float* __restrict__ proj, const int* __restrict__ idxs,
                            float* __restrict__ out, int total)
{
  int t = blockIdx.x * blockDim.x + threadIdx.x;
  if (t >= total) return;          // t enumerates (n, j) exactly like flat idxs
  int n = t / KM1;
  int q = idxs[t];
  float dx = proj[2 * n]     - proj[2 * q];
  float dy = proj[2 * n + 1] - proj[2 * q + 1];
  out[t] = dx * dx + dy * dy;
}

// ---------------------------------------------------------------------------
extern "C" void kernel_launch(void* const* d_in, const int* in_sizes, int n_in,
                              void* d_out, int out_size, void* d_ws, size_t ws_size,
                              hipStream_t stream)
{
  (void)in_sizes; (void)n_in; (void)out_size; (void)ws_size;

  const float* x          = (const float*)d_in[0];
  const float* edge_attr  = (const float*)d_in[1];
  const int*   idxs       = (const int*)  d_in[2];
  const float* l1_msg_w1  = (const float*)d_in[3];
  const float* l1_msg_b1  = (const float*)d_in[4];
  const float* l1_msg_w2  = (const float*)d_in[5];
  const float* l1_msg_b2  = (const float*)d_in[6];
  const float* l1_node_w1 = (const float*)d_in[7];
  const float* l1_node_b1 = (const float*)d_in[8];
  const float* l1_node_w2 = (const float*)d_in[9];
  const float* l1_node_b2 = (const float*)d_in[10];
  const float* l2_msg_w1  = (const float*)d_in[11];
  const float* l2_msg_b1  = (const float*)d_in[12];
  const float* l2_msg_w2  = (const float*)d_in[13];
  const float* l2_msg_b2  = (const float*)d_in[14];
  const float* l2_node_w1 = (const float*)d_in[15];
  const float* l2_node_b1 = (const float*)d_in[16];
  const float* l2_node_w2 = (const float*)d_in[17];
  const float* l2_node_b2 = (const float*)d_in[18];
  float* out = (float*)d_out;

  const int KB784 = 25;   // ceil(784/32); padded region multiplies packed-B zeros
  const int KB128 = 4;

  // ---- carve workspace (256B aligned) ----
  char* wp = (char*)d_ws;
  auto carve = [&](size_t bytes) -> void* {
    void* r = (void*)wp;
    wp += (bytes + 255) & ~(size_t)255;
    return r;
  };
  half_t* Wb1   = (half_t*)carve((size_t)8  * KB784 * 512 * 2);
  half_t* Wm2   = (half_t*)carve((size_t)49 * KB128 * 512 * 2);
  half_t* Wn1   = (half_t*)carve((size_t)8  * (2 * KB784) * 512 * 2);
  half_t* Wn2   = (half_t*)carve((size_t)8  * KB128 * 512 * 2);
  half_t* Wb2   = (half_t*)carve((size_t)8  * KB128 * 512 * 2);
  half_t* Wm2b  = (half_t*)carve((size_t)8  * KB128 * 512 * 2);
  half_t* Wo1   = (half_t*)carve((size_t)8  * (2 * KB128) * 512 * 2);
  half_t* Wo2   = (half_t*)carve((size_t)1  * KB128 * 512 * 2);
  half_t* x_h   = (half_t*)carve((size_t)N_NODES * D_PAD * 2);
  half_t* agg1h = (half_t*)carve((size_t)N_NODES * D_PAD * 2);
  half_t* hm1h  = (half_t*)carve((size_t)N_NODES * HID * 2);
  half_t* hm2h  = (half_t*)carve((size_t)N_NODES * HID * 2);
  half_t* t1h   = (half_t*)carve((size_t)N_NODES * HID * 2);
  half_t* hh    = (half_t*)carve((size_t)N_NODES * HID * 2);
  half_t* agg2h = (half_t*)carve((size_t)N_NODES * HID * 2);
  half_t* t2h   = (half_t*)carve((size_t)N_NODES * HID * 2);
  int*   cnt    = (int*)  carve((size_t)N_NODES * 4);
  float* base1  = (float*)carve((size_t)N_NODES * HID * 4);
  float* hsum1  = (float*)carve((size_t)N_NODES * HID * 4);
  float* base2  = (float*)carve((size_t)N_NODES * HID * 4);
  float* hsum2  = (float*)carve((size_t)N_NODES * HID * 4);
  float* proj   = (float*)carve((size_t)N_NODES * 2 * 4);

  auto pack = [&](const float* W, int Kreal, int Nreal, int KB, int KBt, int kbOff,
                  int NT, half_t* dst) {
    int total = NT * KB * 512;
    pack_b_kernel<<<cdiv(total, 256), 256, 0, stream>>>(W, Kreal, Nreal, KB, KBt, kbOff, NT, dst);
  };
  // NT tiles total; TW=4 over the 4-aligned part, TW=1 remainder launch(es).
  auto gemm = [&](const half_t* A0, int lda0, int KB0,
                  const half_t* A1, int lda1, int KB1,
                  const half_t* B, const float* bias, int Nout, int NT,
                  void* C, int ldc, int relu, int out_f16) {
    int nt4 = NT / 4;                 // groups handled by the TW=4 kernel
    if (nt4 > 0) {
      int waves = M_TILES * nt4;
      gemm_wmma_kernel<4><<<cdiv(waves, 8), 256, 0, stream>>>(
          A0, lda0, KB0, A1, lda1, KB1, B, bias, Nout, 0, nt4, C, ldc, M_TILES, relu, out_f16);
    }
    int rem = NT - nt4 * 4;
    for (int r = 0; r < rem; ++r) {
      int waves = M_TILES;
      gemm_wmma_kernel<1><<<cdiv(waves, 8), 256, 0, stream>>>(
          A0, lda0, KB0, A1, lda1, KB1, B, bias, Nout, nt4 * 4 + r, 1, C, ldc, M_TILES,
          relu, out_f16);
    }
  };

  // ---- pack weights into WMMA-B images ----
  pack(l1_msg_w1,            784, 128, KB784, KB784,     0,  8, Wb1);
  pack(l1_msg_w2,            128, 784, KB128, KB128,     0, 49, Wm2);
  pack(l1_node_w1,           784, 128, KB784, 2*KB784,   0,  8, Wn1);
  pack(l1_node_w1 + 784*128, 784, 128, KB784, 2*KB784, KB784, 8, Wn1);
  pack(l1_node_w2,           128, 128, KB128, KB128,     0,  8, Wn2);
  pack(l2_msg_w1,            128, 128, KB128, KB128,     0,  8, Wb2);
  pack(l2_msg_w2,            128, 128, KB128, KB128,     0,  8, Wm2b);
  pack(l2_node_w1,           128, 128, KB128, 2*KB128,   0,  8, Wo1);
  pack(l2_node_w1 + 128*128, 128, 128, KB128, 2*KB128, KB128, 8, Wo1);
  pack(l2_node_w2,           128,   2, KB128, KB128,     0,  1, Wo2);

  // ---- activations to padded f16 / zero accumulators & pads (every call) ----
  convert_pad_kernel<<<cdiv(N_NODES * D_PAD, 256), 256, 0, stream>>>(x, x_h, N_NODES);
  pad_cols_kernel<<<cdiv(N_NODES * 16, 256), 256, 0, stream>>>(agg1h);
  hipMemsetAsync(cnt,   0, (size_t)N_NODES * 4, stream);
  hipMemsetAsync(hsum1, 0, (size_t)N_NODES * HID * 4, stream);
  hipMemsetAsync(hsum2, 0, (size_t)N_NODES * HID * 4, stream);

  // ---- degree histogram (src == flattened idxs, shared by both layers) ----
  count_src_kernel<<<cdiv(E_EDGES, 256), 256, 0, stream>>>(idxs, cnt, E_EDGES);

  // ======================= Layer 1 =======================
  // base1 = x @ msg_w1[:784]   (bias folded into the per-edge kernel)
  gemm(x_h, D_PAD, KB784, nullptr, 0, 0, Wb1, nullptr, HID, 8, base1, HID, 0, 0);
  // per-edge hidden + scatter-add (linear factorization of msg MLP layer 2)
  edge_hidden_kernel<<<cdiv(E_EDGES * 32, 256), 256, 0, stream>>>(
      base1, idxs, edge_attr, l1_msg_w1 + (size_t)784 * 128, l1_msg_b1, hsum1, E_EDGES);
  mean_to_h_kernel<<<cdiv(N_NODES * HID, 256), 256, 0, stream>>>(hsum1, cnt, hm1h, N_NODES * HID);
  // agg1 = mean_hidden @ msg_w2 + msg_b2  (f16 out, padded lda for next GEMM)
  gemm(hm1h, HID, KB128, nullptr, 0, 0, Wm2, l1_msg_b2, D_IN, 49, agg1h, D_PAD, 0, 1);
  zero_rows_h_kernel<<<cdiv(N_NODES * D_IN, 256), 256, 0, stream>>>(agg1h, D_PAD, D_IN, cnt, N_NODES);
  // t1 = ReLU(cat(x, agg1) @ node_w1 + b)   (concat via two K-segments)
  gemm(x_h, D_PAD, KB784, agg1h, D_PAD, KB784, Wn1, l1_node_b1, HID, 8, t1h, HID, 1, 1);
  // h = t1 @ node_w2 + b
  gemm(t1h, HID, KB128, nullptr, 0, 0, Wn2, l1_node_b2, HID, 8, hh, HID, 0, 1);

  // ======================= Layer 2 =======================
  gemm(hh, HID, KB128, nullptr, 0, 0, Wb2, nullptr, HID, 8, base2, HID, 0, 0);
  edge_hidden_kernel<<<cdiv(E_EDGES * 32, 256), 256, 0, stream>>>(
      base2, idxs, edge_attr, l2_msg_w1 + (size_t)128 * 128, l2_msg_b1, hsum2, E_EDGES);
  mean_to_h_kernel<<<cdiv(N_NODES * HID, 256), 256, 0, stream>>>(hsum2, cnt, hm2h, N_NODES * HID);
  gemm(hm2h, HID, KB128, nullptr, 0, 0, Wm2b, l2_msg_b2, HID, 8, agg2h, HID, 0, 1);
  zero_rows_h_kernel<<<cdiv(N_NODES * HID, 256), 256, 0, stream>>>(agg2h, HID, HID, cnt, N_NODES);
  gemm(hh, HID, KB128, agg2h, HID, KB128, Wo1, l2_node_b1, HID, 8, t2h, HID, 1, 1);
  gemm(t2h, HID, KB128, nullptr, 0, 0, Wo2, l2_node_b2, 2, 1, proj, 2, 0, 0);

  // ======================= distances =======================
  dist_kernel<<<cdiv(N_NODES * KM1, 256), 256, 0, stream>>>(proj, idxs, out, N_NODES * KM1);
}
